// LinAngularAttention_4045859193662
// MI455X (gfx1250) — matrix operand order repaired and checked by reading, not compiled, not run on hardware
//
#include <hip/hip_runtime.h>

// ---- CDNA5 (gfx1250) fused LinAngularAttention ---------------------------
// out[l, h*32+w] = (cos(qh,kh) + dconv_w[h,0,4,0]) * (v@Wv^T+bv)[l, h*32+w]
// qh = (q@Wq^T+bq) head slice, kh = (k@Wv^T+bv) head slice.
// Three 16384x256x256 GEMMs via v_wmma_f32_16x16x32_bf16 (f32 accumulate).
// A tiles staged global->LDS with the Tensor Data Mover (TENSORcnt-pipelined);
// weights pre-converted to bf16 in d_ws by a tiny one-shot kernel.

typedef __attribute__((ext_vector_type(16))) __bf16 v16bf;
typedef __attribute__((ext_vector_type(8)))  float  v8f;
typedef __attribute__((ext_vector_type(4)))  float  v4f;
typedef __attribute__((ext_vector_type(4)))  unsigned int u32x4;
typedef __attribute__((ext_vector_type(4)))  int    i32x4;
typedef __attribute__((ext_vector_type(8)))  int    i32x8;
typedef __attribute__((address_space(3)))    void   lds_void;

#define CDIM   256
#define TILE_M 16
#define NHEAD  8
#define DHEAD  32
#define KS     9

#if defined(__AMDGCN__) && __has_builtin(__builtin_amdgcn_tensor_load_to_lds)
#define USE_TDM 1
#else
#define USE_TDM 0
#endif

// ---------------- fragment builders (contiguous k order, k = 16*h + e) ------------
// A (16x32 bf16): row m = lane&15; koff = (lane>=16)?16:0
//   elem e = A[m][k0+koff+e], e = 0..15 (16 consecutive)
__device__ __forceinline__ v16bf frag_a_f32(const float* __restrict__ arow,
                                            int k0, int koff) {
    const float* p = arow + k0 + koff;
    v4f a0 = *(const v4f*)(p);
    v4f a1 = *(const v4f*)(p + 4);
    v4f a2 = *(const v4f*)(p + 8);
    v4f a3 = *(const v4f*)(p + 12);
    v16bf f;
#pragma unroll
    for (int e = 0; e < 4; ++e) {
        f[e]      = (__bf16)a0[e];
        f[4 + e]  = (__bf16)a1[e];
        f[8 + e]  = (__bf16)a2[e];
        f[12 + e] = (__bf16)a3[e];
    }
    return f;
}

// B (32x16 bf16, KxN) for Out = A @ W^T: col n = lane&15, kb = (lane>=16)?16:0
//   elem e = W[n0+n][k0+kb+e], e = 0..15 (16 consecutive)
__device__ __forceinline__ v16bf frag_b_f32(const float* __restrict__ wrow,
                                            int k0) {
    const float* p = wrow + k0;
    v4f b0 = *(const v4f*)(p);
    v4f b1 = *(const v4f*)(p + 4);
    v4f b2 = *(const v4f*)(p + 8);
    v4f b3 = *(const v4f*)(p + 12);
    v16bf f;
#pragma unroll
    for (int e = 0; e < 4; ++e) {
        f[e]      = (__bf16)b0[e];
        f[4 + e]  = (__bf16)b1[e];
        f[8 + e]  = (__bf16)b2[e];
        f[12 + e] = (__bf16)b3[e];
    }
    return f;
}

// ---------------- Tensor Data Mover staging -------------------------------
#if USE_TDM
// 2D tile load: 16 rows x 256 f32, row stride 256 f32, global -> LDS.
__device__ __forceinline__ void tdm_load_16x256_f32(const float* gsrc,
                                                    float* ldst) {
    const unsigned long long ga = (unsigned long long)gsrc;
    const unsigned lds_off =
        (unsigned)(unsigned long long)(lds_void*)ldst;

    u32x4 g0;
    g0[0] = 1u;                                        // count=1 (valid D#)
    g0[1] = lds_off;                                   // lds_addr [63:32]
    g0[2] = (unsigned)(ga & 0xffffffffull);            // global_addr lo
    g0[3] = (unsigned)((ga >> 32) & 0x01ffffffull)     // global_addr[56:32]
            | (2u << 30);                              // type = 2 (image)

    i32x8 g1;
    g1[0] = 0x20000;            // data_size=2 (4B); mask/flags/pad = 0
    g1[1] = (CDIM << 16);       // tensor_dim0[15:0]=256 ; atomic_bar_addr=0
    g1[2] = (16384 << 16);      // tensor_dim1[15:0]=16384 ; dim0[31:16]=0
    g1[3] = (CDIM << 16);       // tile_dim0=256 ; dim1[31:16]=0
    g1[4] = TILE_M;             // tile_dim1=16 ; tile_dim2=0
    g1[5] = CDIM;               // tensor_dim0_stride[31:0]=256
    g1[6] = 0;                  // stride0[47:32]=0 ; stride1[15:0]=0
    g1[7] = 0;                  // stride1[47:16]=0 (2D: unused)

    i32x4 g2 = {0, 0, 0, 0};    // groups 2/3 unused for 2D tensor
    i32x4 g3 = {0, 0, 0, 0};
#if __clang_major__ >= 23
    i32x8 g4 = {0, 0, 0, 0, 0, 0, 0, 0};
    __builtin_amdgcn_tensor_load_to_lds(g0, g1, g2, g3, g4, 0);
#else
    __builtin_amdgcn_tensor_load_to_lds(g0, g1, g2, g3, 0);
#endif
}
#endif

// ---------------- one projection: D = A @ W^T + bias ----------------------
// MAT is a template arg so the s_wait_tensorcnt immediate is an ICE.
template <int MAT>
__device__ __forceinline__ void process_mat(float* __restrict__ tilemat,
                                            const float* __restrict__ wf,
                                            const __bf16* __restrict__ wb,
                                            const float* __restrict__ bias,
                                            bool use_wbf,
                                            int nl, int koff, int kb, int rz) {
#if USE_TDM
    __builtin_amdgcn_s_wait_tensorcnt(2 - MAT);   // TDM completes in-order
#endif
    const float* arow = tilemat + nl * CDIM;      // A tile row (LDS, f32)
    v16bf afr[8];
#pragma unroll
    for (int ks = 0; ks < 8; ++ks)
        afr[ks] = frag_a_f32(arow, ks * 32, kb);

    if (use_wbf) {
        // bf16 weights pre-converted in d_ws: straight 32B fragment loads.
        for (int nt = 0; nt < 16; ++nt) {
            const int n0 = nt * 16;
            const __bf16* wrow = wb + (n0 + nl) * CDIM + kb;
            v8f acc = {};
#pragma unroll
            for (int ks = 0; ks < 8; ++ks) {
                v16bf bfr = *(const v16bf*)(wrow + ks * 32);
                acc = __builtin_amdgcn_wmma_f32_16x16x32_bf16(
                          false, afr[ks], false, bfr, (short)0, acc,
                          false, false);
            }
            const float bb = bias[n0 + nl];
#pragma unroll
            for (int r = 0; r < 8; ++r)
                tilemat[(rz + r) * CDIM + n0 + nl] = acc[r] + bb;
        }
    } else {
        // Fallback: convert f32 weights on the fly.
        for (int nt = 0; nt < 16; ++nt) {
            const int n0 = nt * 16;
            const float* wrow = wf + (n0 + nl) * CDIM + kb;
            v8f acc = {};
#pragma unroll
            for (int ks = 0; ks < 8; ++ks) {
                v16bf bfr = frag_b_f32(wrow, ks * 32);
                acc = __builtin_amdgcn_wmma_f32_16x16x32_bf16(
                          false, afr[ks], false, bfr, (short)0, acc,
                          false, false);
            }
            const float bb = bias[n0 + nl];
#pragma unroll
            for (int r = 0; r < 8; ++r)
                tilemat[(rz + r) * CDIM + n0 + nl] = acc[r] + bb;
        }
    }
}

// ---------------- one-shot weight conversion to bf16 ----------------------
__global__ __launch_bounds__(256)
void convert_w_bf16(const float* __restrict__ Wq, const float* __restrict__ Wv,
                    unsigned short* __restrict__ dst) {
    const int i = blockIdx.x * 256 + threadIdx.x;          // 0 .. 65535
    const __bf16 a = (__bf16)Wq[i];
    const __bf16 b = (__bf16)Wv[i];
    dst[i]               = __builtin_bit_cast(unsigned short, a);
    dst[CDIM * CDIM + i] = __builtin_bit_cast(unsigned short, b);
}

// ---------------- fused kernel --------------------------------------------
__global__ __launch_bounds__(32)
void lin_angular_fused(const float* __restrict__ q,
                       const float* __restrict__ k,
                       const float* __restrict__ v,
                       const float* __restrict__ Wq,
                       const float* __restrict__ bq,
                       const float* __restrict__ Wv,
                       const float* __restrict__ bv,
                       const float* __restrict__ dw,
                       const unsigned short* __restrict__ wbf_raw,
                       float* __restrict__ out) {
    // tile[] holds staged A (f32) and is overwritten by the projection tiles.
    __shared__ float tile[3][TILE_M * CDIM];               // 48 KB
    __shared__ float scale_s[TILE_M * NHEAD];              // 512 B

    const int lane = threadIdx.x;                          // one wave32/block
    const int m0   = blockIdx.x * TILE_M;

    const float* Amat[3] = {q, k, v};

    const __bf16* wq_bf = (const __bf16*)wbf_raw;
    const __bf16* wv_bf = wbf_raw ? wq_bf + CDIM * CDIM : nullptr;
    const bool use_wbf  = (wbf_raw != nullptr);

    const int nl   = lane & 15;
    const int hi   = lane >> 4;
    const int koff = hi << 3;          // 0 or 8; unused (A frag uses kb)
    const int kb   = hi << 4;          // B frag: 0 or 16
    const int rz   = hi << 3;          // D frag: row base 0 or 8

    // ---- stage A tiles (q,k,v rows) into LDS ----
#if USE_TDM
#pragma unroll
    for (int t = 0; t < 3; ++t)
        tdm_load_16x256_f32(Amat[t] + m0 * CDIM, &tile[t][0]);
    // waits are pipelined inside process_mat<> (tensorcnt 2,1,0)
#else
    for (int t = 0; t < 3; ++t) {
        const v4f* g = (const v4f*)(Amat[t] + m0 * CDIM);
        v4f* l = (v4f*)&tile[t][0];
        for (int i = lane; i < TILE_M * CDIM / 4; i += 32) l[i] = g[i];
    }
    __syncthreads();
#endif

    // ---- three projections (results overwrite the staged A in-place) ----
    process_mat<0>(&tile[0][0], Wq, wq_bf, bq, use_wbf, nl, koff, kb, rz);
    process_mat<1>(&tile[1][0], Wv, wv_bf, bv, use_wbf, nl, koff, kb, rz);
    process_mat<2>(&tile[2][0], Wv, wv_bf, bv, use_wbf, nl, koff, kb, rz);
    __syncthreads();

    // ---- epilogue 1: per (row, head) cosine + dconv center tap ----
    for (int p = lane; p < TILE_M * NHEAD; p += 32) {
        const int row = p >> 3;
        const int h   = p & 7;
        const float* qr = &tile[0][row * CDIM + h * DHEAD];
        const float* kr = &tile[1][row * CDIM + h * DHEAD];
        float s = 0.f, nq = 0.f, nk = 0.f;
#pragma unroll
        for (int j = 0; j < DHEAD; ++j) {
            const float a = qr[j], b = kr[j];
            s  += a * b;
            nq += a * a;
            nk += b * b;
        }
        scale_s[p] = s * rsqrtf(nq * nk) + dw[h * KS + 4];
    }
    __syncthreads();

    // ---- epilogue 2: coalesced scaled store of the v-projection ----
    for (int idx = lane; idx < TILE_M * CDIM; idx += 32) {
        const int row = idx >> 8;
        const int c   = idx & (CDIM - 1);
        const int h   = c >> 5;
        out[(m0 + row) * CDIM + c] = scale_s[row * NHEAD + h] * tile[2][idx];
    }
}

extern "C" void kernel_launch(void* const* d_in, const int* in_sizes, int n_in,
                              void* d_out, int out_size, void* d_ws, size_t ws_size,
                              hipStream_t stream) {
    const float* q  = (const float*)d_in[0];
    const float* k  = (const float*)d_in[1];
    const float* v  = (const float*)d_in[2];
    const float* Wq = (const float*)d_in[3];
    const float* bq = (const float*)d_in[4];
    const float* Wv = (const float*)d_in[5];
    const float* bv = (const float*)d_in[6];
    const float* dw = (const float*)d_in[7];
    float* out = (float*)d_out;

    // Pre-convert both weight matrices to bf16 in scratch if it fits (256 KB).
    unsigned short* wbf = nullptr;
    const size_t need = (size_t)2 * CDIM * CDIM * sizeof(unsigned short);
    if (ws_size >= need) {
        wbf = (unsigned short*)d_ws;
        convert_w_bf16<<<CDIM * CDIM / 256, 256, 0, stream>>>(Wq, Wv, wbf);
    }

    const int L = in_sizes[0] / CDIM;         // 16384 rows (N==1)
    dim3 grid(L / TILE_M);                    // 1024 tiles of 16 rows
    lin_angular_fused<<<grid, 32, 0, stream>>>(q, k, v, Wq, bq, Wv, bv, dw,
                                               wbf, out);
}
